// NetworkForAGraphWithAttributes_46042049413710
// MI455X (gfx1250) — compile-verified, matrix operand order checked by reading, not run on hardware
//
#include <hip/hip_runtime.h>
#include <hip/hip_bf16.h>
#include <math.h>

typedef __attribute__((ext_vector_type(16))) __bf16 v16bf;
typedef __attribute__((ext_vector_type(8)))  float  v8f;

#define N_NODES 20000
#define N_EDGES 320000
#define FEAT    128
#define A_DIM   13      // 4 edge_attr + 9 SH
#define NB      10
#define FC      100
#define KDIM    (FEAT * A_DIM)   // 1664
#define KC_TOT  (KDIM / 32)      // 52
#define KWIN    13               // k-chunk window staged in LDS (4 windows)
#define KW2     4                // k-chunks for W2 GEMM (FC=100 padded to 128)
#define GRAPHS  16
#define OUTF    64

// ---------------------------------------------------------------------------
// 1) Edge geometry: spherical harmonics + radial embedding
// ---------------------------------------------------------------------------
__global__ void geo_kernel(const float* __restrict__ pos,
                           const float* __restrict__ edge_attr,
                           const int*   __restrict__ eidx,
                           float* __restrict__ ea,   // E x 13
                           float* __restrict__ emb)  // E x 10
{
    int e = blockIdx.x * blockDim.x + threadIdx.x;
    if (e >= N_EDGES) return;
    unsigned s = (unsigned)eidx[e];
    unsigned d = (unsigned)eidx[N_EDGES + e];
    float vx = pos[s*3+0] - pos[d*3+0];
    float vy = pos[s*3+1] - pos[d*3+1];
    float vz = pos[s*3+2] - pos[d*3+2];
    float r  = sqrtf(vx*vx + vy*vy + vz*vz);
    float inv = 1.0f / fmaxf(r, 1e-12f);
    float x = vx*inv, y = vy*inv, z = vz*inv;

    const float s3  = 1.7320508075688772f;
    const float s5  = 2.2360679774997896f;
    const float s15 = 3.8729833462074170f;

    float* eap = ea + (size_t)e * A_DIM;
    eap[0] = edge_attr[e*4+0];
    eap[1] = edge_attr[e*4+1];
    eap[2] = edge_attr[e*4+2];
    eap[3] = edge_attr[e*4+3];
    eap[4]  = 1.0f;
    eap[5]  = s3 * x;
    eap[6]  = s3 * y;
    eap[7]  = s3 * z;
    eap[8]  = s15 * x * z;
    eap[9]  = s15 * x * y;
    eap[10] = s5 * (y*y - 0.5f*(x*x + z*z));
    eap[11] = s15 * y * z;
    eap[12] = s15 * 0.5f * (z*z - x*x);

    const float step = 3.5f / 9.0f;
    const float halfpi = 1.5707963267948966f;
    const float sNB = 3.1622776601683795f; // sqrt(10)
    float* ep = emb + (size_t)e * NB;
    #pragma unroll
    for (int b = 0; b < NB; ++b) {
        float diff = (r - (float)b * step) / step;
        ep[b] = (diff > -1.0f && diff < 1.0f) ? cosf(halfpi * diff) * sNB : 0.0f;
    }
}

// ---------------------------------------------------------------------------
// 2) Pack Wtp (fi x 13 x fo, f32) into per-lane bf16 WMMA B-fragments.
//    Bp[((kc*NT + nt)*32 + lane)*16 + j]; lane group g=lane>>4,
//    j<8 -> k=8g+j, else k=8+8g+j (ISA 16-bit operand layout).
// ---------------------------------------------------------------------------
__global__ void packb_kernel(const float* __restrict__ Wtp,
                             __bf16* __restrict__ Bp, int fo, int NT)
{
    int cell = blockIdx.x * blockDim.x + threadIdx.x;
    int ncell = KC_TOT * NT * 32;
    if (cell >= ncell) return;
    int lane = cell & 31;
    int rest = cell >> 5;
    int nt = rest % NT;
    int kc = rest / NT;
    int n = nt * 16 + (lane & 15);
    int g = lane >> 4;
    __bf16* out = Bp + (size_t)cell * 16;
    #pragma unroll
    for (int j = 0; j < 16; ++j) {
        int ki = (j < 8) ? (8*g + j) : (8 + 8*g + j);
        int k  = kc * 32 + ki;
        int i  = k & 127;       // fi index
        int a  = k >> 7;        // attr index
        out[j] = (__bf16)Wtp[((size_t)i * A_DIM + a) * fo + n];
    }
}

// ---------------------------------------------------------------------------
// 2b) Pack W2 (100 x 128, f32) into bf16 B-fragments, K padded 100 -> 128.
//     Wp[((kc*8 + nt)*32 + lane)*16 + j]
// ---------------------------------------------------------------------------
__global__ void packw2_kernel(const float* __restrict__ W2,
                              __bf16* __restrict__ Wp)
{
    int cell = blockIdx.x * blockDim.x + threadIdx.x;
    if (cell >= KW2 * 8 * 32) return;
    int lane = cell & 31;
    int rest = cell >> 5;
    int nt = rest & 7;
    int kc = rest >> 3;
    int n = nt * 16 + (lane & 15);
    int g = lane >> 4;
    __bf16* out = Wp + (size_t)cell * 16;
    #pragma unroll
    for (int j = 0; j < 16; ++j) {
        int ki = (j < 8) ? (8*g + j) : (8 + 8*g + j);
        int k  = kc * 32 + ki;
        out[j] = (__bf16)((k < FC) ? W2[k * FEAT + n] : 0.0f);
    }
}

// ---------------------------------------------------------------------------
// 3) Fused message kernel. Block = 16 edges, 256 threads = 8 waves.
//    Stage 1: h = relu(emb@W1)            (VALU, 10x100 tiny)
//    Stage 2: w = h@W2 via 4 WMMA/wave    (K padded to 128)
//             xg = x[src]*w written from D-tile layout
//    Stage 3: A = ea (x) xg fragments in LDS, 52 WMMA/wave vs packed Wtp
//    Stage 4: atomic scatter of D tiles into agg[dst].
//    All global gathers/scatters use unsigned 32-bit element offsets so the
//    backend can use SGPR-base + 32-bit VGPR-offset addressing.
// ---------------------------------------------------------------------------
__global__ __launch_bounds__(256)
void msg_kernel(const float* __restrict__ x,
                const float* __restrict__ ea,
                const float* __restrict__ emb,
                const int*   __restrict__ eidx,
                const float* __restrict__ W1,    // 10 x 100
                const __bf16* __restrict__ W2p,  // packed W2
                const __bf16* __restrict__ Bp,   // packed Wtp
                float* __restrict__ agg,         // N x fo
                int fo, int NT)
{
    __shared__ float    s_emb[16][NB];
    __shared__ float    s_ea [16][A_DIM];
    __shared__ float    s_h  [16][FC];
    __shared__ float    s_xg [16][FEAT];
    __shared__ unsigned s_srcrow[16];   // src * FEAT (element offset)
    __shared__ unsigned s_dstrow[16];   // dst * fo   (element offset)
    __shared__ __attribute__((aligned(32))) __bf16 s_Ah[KW2 * 32 * 16];
    __shared__ __attribute__((aligned(32))) __bf16 s_A [KWIN * 32 * 16];

    const int t   = threadIdx.x;
    const int el  = t >> 4;
    const int sub = t & 15;
    const int e   = blockIdx.x * 16 + el;

    if (sub == 0) {
        s_srcrow[el] = (unsigned)eidx[e] * (unsigned)FEAT;
        s_dstrow[el] = (unsigned)eidx[N_EDGES + e] * (unsigned)fo;
    }
    if (sub < NB)    s_emb[el][sub] = emb[(size_t)e * NB + sub];
    if (sub < A_DIM) s_ea [el][sub] = ea [(size_t)e * A_DIM + sub];
    __syncthreads();

    // Stage 1: hidden = relu(emb @ W1)
    for (int c = sub; c < FC; c += 16) {
        float acc = 0.0f;
        #pragma unroll
        for (int b = 0; b < NB; ++b) acc += s_emb[el][b] * W1[b * FC + c];
        s_h[el][c] = fmaxf(acc, 0.0f);
    }
    __syncthreads();

    // Stage 2a: build h fragments (K padded 100 -> 128)
    for (int cell = t; cell < KW2 * 32; cell += 256) {
        int ln = cell & 31;
        int kc = cell >> 5;
        int m  = ln & 15;
        int g  = ln >> 4;
        v16bf frag;
        #pragma unroll
        for (int j = 0; j < 16; ++j) {
            int ki = (j < 8) ? (8*g + j) : (8 + 8*g + j);
            int k  = kc * 32 + ki;
            frag[j] = (__bf16)((k < FC) ? s_h[m][k] : 0.0f);
        }
        *(v16bf*)&s_Ah[(size_t)cell * 16] = frag;
    }
    __syncthreads();

    const int lane = t & 31;
    const int wid  = t >> 5;

    // Stage 2b: w tile via WMMA (all 8 waves; fi = 128 always), then xg
    {
        v8f wacc = {};
        #pragma unroll
        for (int kc = 0; kc < KW2; ++kc) {
            v16bf a = *(const v16bf*)&s_Ah[((size_t)kc * 32 + lane) * 16];
            v16bf b = *(const v16bf*)(W2p + (((size_t)kc * 8 + wid) * 32 + lane) * 16);
            wacc = __builtin_amdgcn_wmma_f32_16x16x32_bf16(
                     false, a, false, b, (short)0, wacc, false, false);
        }
        unsigned col   = (unsigned)(wid * 16 + (lane & 15));
        int      rbase = (lane >> 4) * 8;
        #pragma unroll
        for (int v = 0; v < 8; ++v) {
            int m = rbase + v;
            s_xg[m][col] = x[s_srcrow[m] + col] * wacc[v];
        }
    }

    // Stage 3: tensor-product GEMM over K = 1664 in 4 windows of 13 chunks
    v8f acc = {};
    for (int kb = 0; kb < KC_TOT / KWIN; ++kb) {
        const int kc0 = kb * KWIN;
        __syncthreads();   // s_xg ready / previous window consumed
        // A[e,k] = ea[e, k>>7] * xg[e, k&127]; a-index constant per 32-chunk
        for (int cell = t; cell < KWIN * 32; cell += 256) {
            int ln = cell & 31;
            int kk = cell >> 5;
            int m  = ln & 15;
            int g  = ln >> 4;
            int kbase = (kc0 + kk) * 32;
            float eav = s_ea[m][kbase >> 7];
            const float* xr = &s_xg[m][(kbase & 127) + 8*g];
            v16bf frag;
            #pragma unroll
            for (int j = 0; j < 8; ++j) frag[j]     = (__bf16)(eav * xr[j]);
            #pragma unroll
            for (int j = 0; j < 8; ++j) frag[8 + j] = (__bf16)(eav * xr[16 + j]);
            *(v16bf*)&s_A[(size_t)cell * 16] = frag;
        }
        __syncthreads();
        if (wid < NT) {
            for (int kk = 0; kk < KWIN; ++kk) {
                v16bf a = *(const v16bf*)&s_A[((size_t)kk * 32 + lane) * 16];
                int kc = kc0 + kk;
                v16bf b = *(const v16bf*)(Bp + (((size_t)kc * NT + wid) * 32 + lane) * 16);
                acc = __builtin_amdgcn_wmma_f32_16x16x32_bf16(
                        false, a, false, b, (short)0, acc, false, false);
            }
        }
    }

    // Stage 4: scatter D tile (VGPR v: lanes 0-15 -> row v, lanes 16-31 -> 8+v)
    if (wid < NT) {
        unsigned col   = (unsigned)(wid * 16 + (lane & 15));
        int      rbase = (lane >> 4) * 8;
        #pragma unroll
        for (int v = 0; v < 8; ++v) {
            unsigned off = s_dstrow[rbase + v] + col;
            atomicAdd(&agg[off], acc[v]);
        }
    }
}

// ---------------------------------------------------------------------------
// 4) Node update: x' = agg/sqrt(16) + x@Wskip + node_attr@Wna (+ReLU)
// ---------------------------------------------------------------------------
__global__ __launch_bounds__(128)
void update_kernel(const float* __restrict__ xin,
                   const float* __restrict__ agg,
                   const float* __restrict__ na,
                   const float* __restrict__ Wskip,  // fi x fo
                   const float* __restrict__ Wna,    // 8 x fo
                   float* __restrict__ xout,
                   int fo, int do_relu)
{
    __shared__ float sx[FEAT];
    __shared__ float sna[8];
    unsigned n = blockIdx.x;
    int t = threadIdx.x;
    sx[t] = xin[n * FEAT + t];
    if (t < 8) sna[t] = na[n * 8u + t];
    __syncthreads();
    if (t < fo) {
        float acc = agg[n * (unsigned)fo + t] * 0.25f;  // 1/sqrt(16)
        for (int i = 0; i < FEAT; ++i) acc += sx[i] * Wskip[i * fo + t];
        #pragma unroll
        for (int j = 0; j < 8; ++j) acc += sna[j] * Wna[j * fo + t];
        if (do_relu) acc = fmaxf(acc, 0.0f);
        xout[n * (unsigned)fo + t] = acc;
    }
}

// ---------------------------------------------------------------------------
// 5) Batch pooling: out[batch[n]] += x[n] / sqrt(N)
// ---------------------------------------------------------------------------
__global__ void pool_kernel(const float* __restrict__ x,
                            const int* __restrict__ batch,
                            float* __restrict__ out)
{
    unsigned idx = blockIdx.x * blockDim.x + threadIdx.x;
    if (idx >= N_NODES * OUTF) return;
    unsigned n = idx >> 6;
    unsigned o = idx & 63;
    atomicAdd(&out[(unsigned)batch[n] * OUTF + o], x[idx] * 0.007071067811865475f);
}

// ---------------------------------------------------------------------------
extern "C" void kernel_launch(void* const* d_in, const int* in_sizes, int n_in,
                              void* d_out, int out_size, void* d_ws, size_t ws_size,
                              hipStream_t stream)
{
    const float* pos        = (const float*)d_in[0];
    const float* node_input = (const float*)d_in[1];
    const float* node_attr  = (const float*)d_in[2];
    const float* edge_attr  = (const float*)d_in[3];
    const int*   eidx       = (const int*)  d_in[4];
    const int*   batch      = (const int*)  d_in[5];
    // params: 3 layers x {W1, W2, Wtp, Wskip, Wna}
    const float* W1[3], *W2[3], *Wtp[3], *Wskip[3], *Wna[3];
    for (int l = 0; l < 3; ++l) {
        W1[l]    = (const float*)d_in[6 + 5*l + 0];
        W2[l]    = (const float*)d_in[6 + 5*l + 1];
        Wtp[l]   = (const float*)d_in[6 + 5*l + 2];
        Wskip[l] = (const float*)d_in[6 + 5*l + 3];
        Wna[l]   = (const float*)d_in[6 + 5*l + 4];
    }

    // workspace layout (all offsets 256B aligned)
    char* ws = (char*)d_ws;
    float*  ea     = (float*) (ws + 0);                        // E*13*4 = 16,640,000
    float*  emb    = (float*) (ws + 16640000);                 // E*10*4 = 12,800,000
    float*  x0     = (float*) (ws + 29440000);                 // N*128*4 = 10,240,000
    float*  x1     = (float*) (ws + 39680000);                 // N*128*4
    float*  agg    = (float*) (ws + 49920000);                 // N*128*4
    __bf16* Bpack  = (__bf16*)(ws + 60160000);                 // 52*8*32*16*2 = 425,984
    __bf16* W2pack = (__bf16*)(ws + 60585984);                 // 4*8*32*16*2 = 32,768

    // 1) edge geometry
    geo_kernel<<<(N_EDGES + 255) / 256, 256, 0, stream>>>(pos, edge_attr, eidx, ea, emb);

    const int fo_l[3]   = {FEAT, FEAT, OUTF};
    const float* xin_l[3]  = {node_input, x0, x1};
    float*       xout_l[3] = {x0, x1, x0};

    for (int l = 0; l < 3; ++l) {
        int fo = fo_l[l];
        int NT = fo / 16;
        hipMemsetAsync(agg, 0, (size_t)N_NODES * fo * sizeof(float), stream);
        int ncell = KC_TOT * NT * 32;
        packb_kernel<<<(ncell + 255) / 256, 256, 0, stream>>>(Wtp[l], Bpack, fo, NT);
        packw2_kernel<<<(KW2 * 8 * 32 + 255) / 256, 256, 0, stream>>>(W2[l], W2pack);
        msg_kernel<<<N_EDGES / 16, 256, 0, stream>>>(
            xin_l[l], ea, emb, eidx, W1[l], W2pack, Bpack, agg, fo, NT);
        update_kernel<<<N_NODES, 128, 0, stream>>>(
            xin_l[l], agg, node_attr, Wskip[l], Wna[l], xout_l[l], fo, (l < 2) ? 1 : 0);
    }

    hipMemsetAsync(d_out, 0, (size_t)GRAPHS * OUTF * sizeof(float), stream);
    pool_kernel<<<(N_NODES * OUTF + 255) / 256, 256, 0, stream>>>(x0, batch, (float*)d_out);
}
